// GCN_LPA_65910568124787
// MI455X (gfx1250) — matrix operand order repaired and checked by reading, not compiled, not run on hardware
//
#include <hip/hip_runtime.h>
#include <hip/hip_bf16.h>

constexpr int kN  = 50000;
constexpr int kE  = 800000;
constexpr int kIn = 256;
constexpr int kH  = 128;
constexpr int kC  = 40;
constexpr int kCp = 48;   // C padded to 3 WMMA tiles

typedef __attribute__((ext_vector_type(16))) __bf16 v16bf;
typedef __attribute__((ext_vector_type(8)))  float  v8f;
typedef __attribute__((ext_vector_type(4)))  unsigned int v4u;
typedef __attribute__((ext_vector_type(8)))  int    v8i;
typedef __attribute__((ext_vector_type(4)))  int    v4i;

#if defined(__HIP_DEVICE_COMPILE__) && __has_builtin(__builtin_amdgcn_tensor_load_to_lds)
#define HAVE_TDM 1
#else
#define HAVE_TDM 0
#endif

// ---------------- utility kernels ----------------

__global__ void zero_f(float* __restrict__ p, long n) {
  long i = (long)blockIdx.x * blockDim.x + threadIdx.x;
  if (i < n) p[i] = 0.f;
}

__global__ void deg_kernel(const int* __restrict__ src, const int* __restrict__ dst,
                           float* __restrict__ ns, float* __restrict__ nd) {
  int e = blockIdx.x * blockDim.x + threadIdx.x;
  if (e < kE) {
    atomicAdd(&ns[src[e]], 1.f);
    atomicAdd(&nd[dst[e]], 1.f);
  }
}

__global__ void norm_kernel(float* __restrict__ ns, float* __restrict__ nd) {
  int i = blockIdx.x * blockDim.x + threadIdx.x;
  if (i < kN) {
    ns[i] = rsqrtf(fmaxf(ns[i], 1.f));
    nd[i] = rsqrtf(fmaxf(nd[i], 1.f));
  }
}

// ---------------- GEMM 1: h1 = bf16( (X @ W1) * norm_src ) ----------------
// block = 256 threads = 8 waves; each wave computes one 16-row strip x 128 cols.
// W1 K-slab staged via TDM (tensor_load_to_lds) into raw fp32 LDS, then swizzled
// into fragment-native bf16 layout so each B fragment is two ds_load_b128s.

__global__ __launch_bounds__(256) void gemm1_kernel(
    const float* __restrict__ X, const float* __restrict__ W1,
    const float* __restrict__ ns, __bf16* __restrict__ h1)
{
  __shared__ float raw[32 * kH];                                   // 16 KB raw fp32 slab
  __shared__ __attribute__((aligned(32))) __bf16 frag[8 * 32 * 16]; // 8 KB fragment layout

  const int tid  = threadIdx.x;
  const int wave = (blockIdx.x * 256 + tid) >> 5;      // wave-uniform
  const int nTiles = kN / 16;                          // 3125 exact
  const int rt   = wave < nTiles ? wave : nTiles - 1;  // clamp (no early return: barriers+WMMA)
  const int lane = tid & 31;
  const int half = lane >> 4;
  const int ml   = lane & 15;
  const int row  = rt * 16;

  // swizzle-stage identity for this thread: tile nt = tid>>5, lane ln = tid&31
  const int snt = tid >> 5;
  const int sln = tid & 31;
  const int shf = sln >> 4;
  const int scol = snt * 16 + (sln & 15);

  v8f acc[8];
#pragma unroll
  for (int j = 0; j < 8; ++j)
#pragma unroll
    for (int r = 0; r < 8; ++r) acc[j][r] = 0.f;

  for (int kc = 0; kc < kIn; kc += 32) {
    __syncthreads();   // previous-iteration LDS reads done before overwrite

#if HAVE_TDM
    if (tid < 32) {
      // --- Tensor Data Mover: load 32x128 fp32 slab of W1 into LDS ---
      unsigned long long ga = (unsigned long long)(uintptr_t)(W1 + (size_t)kc * kH);
      unsigned lds_off = (unsigned)(uintptr_t)&raw[0];   // low 32 bits = LDS offset
      v4u g0;
      g0[0] = 1u;                                        // count=1 (valid), user mode
      g0[1] = lds_off;                                   // lds_addr
      g0[2] = (unsigned)(ga & 0xFFFFFFFFu);              // global_addr[31:0]
      g0[3] = (unsigned)((ga >> 32) & 0x1FFFFFFu)        // global_addr[56:32]
            | (2u << 30);                                // type=2 ("image")
      v8i g1;
      g1[0] = (int)(2u << 16);          // workgroup_mask=0, data_size=2 (4B)
      g1[1] = (int)(128u << 16);        // tensor_dim0[15:0] = 128   (bits 63:48)
      g1[2] = (int)(256u << 16);        // tensor_dim1[15:0] = 256   (bits 95:80)
      g1[3] = (int)(128u << 16);        // tile_dim0 = 128           (bits 127:112)
      g1[4] = (int)(32u);               // tile_dim1 = 32            (bits 143:128)
      g1[5] = (int)(128u);              // tensor_dim0_stride = 128  (bits 191:160)
      g1[6] = 0;
      g1[7] = 0;
      v4i gz4 = {0, 0, 0, 0};
#if defined(__clang_major__) && __clang_major__ >= 23
      v8i gz8 = {0, 0, 0, 0, 0, 0, 0, 0};
      __builtin_amdgcn_tensor_load_to_lds(g0, g1, gz4, gz4, gz8, 0);
#else
      __builtin_amdgcn_tensor_load_to_lds(g0, g1, gz4, gz4, 0);
#endif
      __builtin_amdgcn_s_wait_tensorcnt(0);
    }
#else
    {
      const float4* wsrc = (const float4*)(W1 + (size_t)kc * kH);
      float4* wdst = (float4*)raw;
#pragma unroll
      for (int t = tid; t < (32 * kH) / 4; t += 256) wdst[t] = wsrc[t];
    }
#endif
    __syncthreads();

    // swizzle raw fp32 -> bf16 fragment layout: frag[(nt*32+ln)*16 + i]
    {
      v16bf tmp;
#pragma unroll
      for (int i = 0; i < 16; ++i)
        tmp[i] = (__bf16)raw[(i + 8 * shf + 8 * (i >> 3)) * kH + scol];
      *(v16bf*)&frag[tid * 16] = tmp;
    }
    __syncthreads();

    // A fragment: row = row+ml, slot i -> K = i + 8*((i>>3)+half)
    const float* arow = X + (size_t)(row + ml) * kIn + kc;
    v16bf a;
#pragma unroll
    for (int i = 0; i < 8; ++i)  a[i] = (__bf16)arow[i + 8 * half];
#pragma unroll
    for (int i = 8; i < 16; ++i) a[i] = (__bf16)arow[i + 8 + 8 * half];

#pragma unroll
    for (int nt = 0; nt < 8; ++nt) {
      v16bf b = *(const v16bf*)&frag[(nt * 32 + lane) * 16];  // 2x ds_load_b128
      acc[nt] = __builtin_amdgcn_wmma_f32_16x16x32_bf16(
          false, a, false, b, (short)0, acc[nt], false, false);
    }
  }

  // D layout: lane -> col = ml; VGPR r -> row = row + r + 8*half
#pragma unroll
  for (int nt = 0; nt < 8; ++nt)
#pragma unroll
    for (int r = 0; r < 8; ++r) {
      int m = row + r + 8 * half;
      int n = nt * 16 + ml;
      h1[(size_t)m * kH + n] = (__bf16)(acc[nt][r] * ns[m]);
    }
}

// ---------------- scatter 1: agg1[dst] += h1[src]  (128 feats/edge) ----------------

__global__ void scatter1_kernel(const int* __restrict__ src, const int* __restrict__ dst,
                                const __bf16* __restrict__ h1, float* __restrict__ agg1)
{
  long t = (long)blockIdx.x * 256 + threadIdx.x;
  if (t >= (long)kE * 32) return;
  int e = (int)(t >> 5);
  int f = (int)(t & 31) * 4;
  int s = src[e], d = dst[e];
  const __bf16* hp = h1 + (size_t)s * kH + f;
  float* ap = agg1 + (size_t)d * kH + f;
#pragma unroll
  for (int i = 0; i < 4; ++i) atomicAdd(&ap[i], (float)hp[i]);
}

// ---------------- GEMM 2: h2 = ( relu(agg1*norm_dst + b1) @ W2 ) * norm_src ----------------
// W2 (128x40, zero-padded to 48) pre-swizzled once into fragment-native LDS layout.

__global__ __launch_bounds__(256) void gemm2_kernel(
    const float* __restrict__ agg1, const float* __restrict__ W2,
    const float* __restrict__ b1, const float* __restrict__ ns,
    const float* __restrict__ nd, float* __restrict__ h2)
{
  __shared__ __attribute__((aligned(32))) __bf16 frag[4 * 3 * 32 * 16]; // 12 KB
  const int tid = threadIdx.x;

  // stage: index t over (kb in 0..3) x (nt in 0..2) x (ln in 0..31) = 384
  for (int t = tid; t < 384; t += 256) {
    int kb  = t / 96;
    int rem = t % 96;
    int nt  = rem >> 5;
    int ln  = rem & 31;
    int hf  = ln >> 4;
    int c   = nt * 16 + (ln & 15);
    v16bf tmp;
#pragma unroll
    for (int i = 0; i < 16; ++i) {
      int krow = kb * 32 + i + 8 * hf + 8 * (i >> 3);
      tmp[i] = (__bf16)((c < kC) ? W2[krow * kC + c] : 0.f);
    }
    *(v16bf*)&frag[t * 16] = tmp;
  }
  __syncthreads();

  const int wave = (blockIdx.x * 256 + tid) >> 5;
  const int nTiles = kN / 16;
  const int rt   = wave < nTiles ? wave : nTiles - 1;
  const int lane = tid & 31;
  const int half = lane >> 4;
  const int ml   = lane & 15;
  const int row  = rt * 16;
  const int m    = row + ml;
  const float ndv = nd[m];

  v8f acc[3];
#pragma unroll
  for (int j = 0; j < 3; ++j)
#pragma unroll
    for (int r = 0; r < 8; ++r) acc[j][r] = 0.f;

  for (int kb = 0; kb < 4; ++kb) {
    const int kc = kb * 32;
    const float* arow = agg1 + (size_t)m * kH + kc;
    v16bf a;
#pragma unroll
    for (int i = 0; i < 8; ++i) {
      int K = i + 8 * half;
      a[i] = (__bf16)fmaxf(arow[K] * ndv + b1[kc + K], 0.f);   // fused norm+bias+relu
    }
#pragma unroll
    for (int i = 8; i < 16; ++i) {
      int K = i + 8 + 8 * half;
      a[i] = (__bf16)fmaxf(arow[K] * ndv + b1[kc + K], 0.f);
    }
#pragma unroll
    for (int nt = 0; nt < 3; ++nt) {
      v16bf b = *(const v16bf*)&frag[((kb * 3 + nt) * 32 + lane) * 16];
      acc[nt] = __builtin_amdgcn_wmma_f32_16x16x32_bf16(
          false, a, false, b, (short)0, acc[nt], false, false);
    }
  }

#pragma unroll
  for (int nt = 0; nt < 3; ++nt)
#pragma unroll
    for (int r = 0; r < 8; ++r) {
      int mm = row + r + 8 * half;
      int n  = nt * 16 + ml;
      if (n < kC) h2[(size_t)mm * kC + n] = acc[nt][r] * ns[mm];
    }
}

// ---------------- scatter 2: outx[dst] += h2[src]  (40 feats/edge) ----------------

__global__ void scatter2_kernel(const int* __restrict__ src, const int* __restrict__ dst,
                                const float* __restrict__ h2, float* __restrict__ outx)
{
  long t = (long)blockIdx.x * 256 + threadIdx.x;
  if (t >= (long)kE * 8) return;
  int e = (int)(t >> 3);
  int f = (int)(t & 7) * 5;
  int s = src[e], d = dst[e];
  const float* hp = h2 + (size_t)s * kC + f;
  float* ap = outx + (size_t)d * kC + f;
#pragma unroll
  for (int i = 0; i < 5; ++i) atomicAdd(&ap[i], hp[i]);
}

__global__ void finx_kernel(float* __restrict__ outx, const float* __restrict__ nd,
                            const float* __restrict__ b2) {
  long t = (long)blockIdx.x * blockDim.x + threadIdx.x;
  if (t < (long)kN * kC) {
    int i = (int)(t / kC), c = (int)(t % kC);
    outx[t] = outx[t] * nd[i] + b2[c];
  }
}

// ---------------- LPA ----------------

__global__ void inity_kernel(const float* __restrict__ labels, const int* __restrict__ mask,
                             float* __restrict__ y) {
  long t = (long)blockIdx.x * blockDim.x + threadIdx.x;
  if (t < (long)kN * kC) y[t] = mask[t / kC] ? labels[t] : 0.f;
}

// y_next[src] += y_cur[dst]   (reference: segment_sum(y[dst], src))
__global__ void lpa_scatter_kernel(const int* __restrict__ src, const int* __restrict__ dst,
                                   const float* __restrict__ ycur, float* __restrict__ ynxt) {
  long t = (long)blockIdx.x * 256 + threadIdx.x;
  if (t >= (long)kE * 8) return;
  int e = (int)(t >> 3);
  int f = (int)(t & 7) * 5;
  int s = src[e], d = dst[e];
  const float* yp = ycur + (size_t)d * kC + f;
  float* ap = ynxt + (size_t)s * kC + f;
#pragma unroll
  for (int i = 0; i < 5; ++i) atomicAdd(&ap[i], yp[i]);
}

__global__ void lpa_mask_kernel(const float* __restrict__ labels, const int* __restrict__ mask,
                                float* __restrict__ y) {
  long t = (long)blockIdx.x * blockDim.x + threadIdx.x;
  if (t < (long)kN * kC && mask[t / kC]) y[t] = labels[t];
}

__global__ void copy_kernel(const float* __restrict__ s, float* __restrict__ d, long n) {
  long i = (long)blockIdx.x * blockDim.x + threadIdx.x;
  if (i < n) d[i] = s[i];
}

// ---------------- host ----------------

static inline int nblk(long n, int bs) { return (int)((n + bs - 1) / bs); }

extern "C" void kernel_launch(void* const* d_in, const int* in_sizes, int n_in,
                              void* d_out, int out_size, void* d_ws, size_t ws_size,
                              hipStream_t stream) {
  const float* X      = (const float*)d_in[0];
  const float* labels = (const float*)d_in[1];
  const int*   mask   = (const int*)d_in[2];
  const int*   src    = (const int*)d_in[3];
  const int*   dst    = (const int*)d_in[4];
  const float* W1     = (const float*)d_in[5];
  const float* b1     = (const float*)d_in[6];
  const float* W2     = (const float*)d_in[7];
  const float* b2     = (const float*)d_in[8];

  float* outx = (float*)d_out;
  float* outy = outx + (size_t)kN * kC;

  // workspace layout (floats): ns[N] nd[N] agg1[128N] h2[40N] yA[40N] yB[40N] h1(bf16=64N)
  float*  ws   = (float*)d_ws;
  float*  ns   = ws;
  float*  nd   = ws + (size_t)kN;
  float*  agg1 = ws + (size_t)2 * kN;
  float*  h2   = ws + (size_t)130 * kN;
  float*  yA   = ws + (size_t)170 * kN;
  float*  yB   = ws + (size_t)210 * kN;
  __bf16* h1   = (__bf16*)(ws + (size_t)250 * kN);

  const int gemmBlocks = nblk((long)(kN / 16) * 32, 256);   // 3125 waves

  // degrees -> norms
  zero_f<<<nblk(2L * kN, 256), 256, 0, stream>>>(ns, 2L * kN);
  deg_kernel<<<nblk(kE, 256), 256, 0, stream>>>(src, dst, ns, nd);
  norm_kernel<<<nblk(kN, 256), 256, 0, stream>>>(ns, nd);

  // layer 1
  zero_f<<<nblk((long)kN * kH, 256), 256, 0, stream>>>(agg1, (long)kN * kH);
  gemm1_kernel<<<gemmBlocks, 256, 0, stream>>>(X, W1, ns, h1);
  scatter1_kernel<<<nblk((long)kE * 32, 256), 256, 0, stream>>>(src, dst, h1, agg1);

  // layer 2 (relu+bias+norm fused into A load)
  gemm2_kernel<<<gemmBlocks, 256, 0, stream>>>(agg1, W2, b1, ns, nd, h2);
  zero_f<<<nblk((long)kN * kC, 256), 256, 0, stream>>>(outx, (long)kN * kC);
  scatter2_kernel<<<nblk((long)kE * 8, 256), 256, 0, stream>>>(src, dst, h2, outx);
  finx_kernel<<<nblk((long)kN * kC, 256), 256, 0, stream>>>(outx, nd, b2);

  // label propagation
  inity_kernel<<<nblk((long)kN * kC, 256), 256, 0, stream>>>(labels, mask, yA);
  float* cur = yA;
  float* nxt = yB;
  for (int it = 0; it < 10; ++it) {
    zero_f<<<nblk((long)kN * kC, 256), 256, 0, stream>>>(nxt, (long)kN * kC);
    lpa_scatter_kernel<<<nblk((long)kE * 8, 256), 256, 0, stream>>>(src, dst, cur, nxt);
    lpa_mask_kernel<<<nblk((long)kN * kC, 256), 256, 0, stream>>>(labels, mask, nxt);
    float* tmp = cur; cur = nxt; nxt = tmp;
  }
  copy_kernel<<<nblk((long)kN * kC, 256), 256, 0, stream>>>(cur, outy, (long)kN * kC);
}
